// MoeBlock_60043642798417
// MI455X (gfx1250) — matrix operand order repaired and checked by reading, not compiled, not run on hardware
//
#include <hip/hip_runtime.h>
#include <hip/hip_bf16.h>
#include <math.h>

#define T_TOK 2048
#define H_DIM 1024
#define F_DIM 4096
#define N_EXP 8
#define ALIGN_M 64                          // M-block: per-expert segments 64-aligned
#define RCAP  (2 * T_TOK + N_EXP * ALIGN_M) // 4608 compact rows
#define MTILES (RCAP / ALIGN_M)             // 72 M-blocks
#define KC    256                           // K staging chunk (bf16 elems)
#define LDP   (KC + 8)                      // padded LDS pitch: 528B = 4 banks -> conflict-free

typedef __attribute__((ext_vector_type(16))) __bf16 v16bf;
typedef __attribute__((ext_vector_type(8)))  float  v8f;
typedef __attribute__((ext_vector_type(4)))  int    v4i;

union Frag {
    v16bf v;
    unsigned int d[8];
    unsigned short u[16];
};

__device__ __forceinline__ unsigned short f2bf(float f) {
    __bf16 h = (__bf16)f;
    return __builtin_bit_cast(unsigned short, h);
}

// ---- gfx1250 async global->LDS staging (ASYNCcnt path), with safe fallback ----
#if __has_builtin(__builtin_amdgcn_global_load_async_to_lds_b128)
#define HAVE_ASYNC_LDS 1
#endif

__device__ __forceinline__ void stage_b128(const void* gsrc, void* ldst) {
#ifdef HAVE_ASYNC_LDS
    // Param types per compiler diagnostic: v4i addrspace(1)* src, v4i addrspace(3)* dst.
    // Integer round-trip: global ptr == 64-bit VA; LDS flat addr low 32 bits are the
    // wave-relative LDS offset (ISA aperture mapping).
    __builtin_amdgcn_global_load_async_to_lds_b128(
        (__attribute__((address_space(1))) v4i*)(unsigned long long)(uintptr_t)gsrc,
        (__attribute__((address_space(3))) v4i*)(unsigned int)(uintptr_t)ldst,
        0, 0);
#else
    *(uint4*)ldst = *(const uint4*)gsrc;
#endif
}

__device__ __forceinline__ void stage_wait() {
#ifdef HAVE_ASYNC_LDS
#if __has_builtin(__builtin_amdgcn_s_wait_asynccnt)
    __builtin_amdgcn_s_wait_asynccnt(0);
#else
    asm volatile("s_wait_asynccnt 0x0" ::: "memory");
#endif
#endif
}

// A fragment (16x32 bf16, M x K) from padded LDS tile [16][ldk].
__device__ __forceinline__ v16bf load_a_frag_lds(const unsigned short* xt, int ldk,
                                                 int kb, int lane) {
    const int row  = lane & 15;
    const int koff = (lane >> 4) * 16;
    const unsigned short* s = xt + row * ldk + kb + koff;
    Frag a;
#pragma unroll
    for (int r = 0; r < 4; ++r) {
        a.d[r]     = *(const unsigned int*)(s + 2 * r);
        a.d[4 + r] = *(const unsigned int*)(s + 8 + 2 * r);
    }
    return a.v;
}

// B fragment (32x16 bf16, K x N) from f32 weights W[n][k] row-major (k contiguous).
__device__ __forceinline__ v16bf load_b_frag_f32(const float* __restrict__ W, int n0,
                                                 long long pitch, int kb, int lane) {
    const int nl = lane & 15;
    const int kh = (lane >> 4) * 16;
    const float* src = W + (long long)(n0 + nl) * pitch + kb + kh;
    float4 f0 = *(const float4*)(src + 0);
    float4 f1 = *(const float4*)(src + 4);
    float4 f2 = *(const float4*)(src + 8);
    float4 f3 = *(const float4*)(src + 12);
    Frag b;
    b.u[0]  = f2bf(f0.x); b.u[1]  = f2bf(f0.y); b.u[2]  = f2bf(f0.z); b.u[3]  = f2bf(f0.w);
    b.u[4]  = f2bf(f1.x); b.u[5]  = f2bf(f1.y); b.u[6]  = f2bf(f1.z); b.u[7]  = f2bf(f1.w);
    b.u[8]  = f2bf(f2.x); b.u[9]  = f2bf(f2.y); b.u[10] = f2bf(f2.z); b.u[11] = f2bf(f2.w);
    b.u[12] = f2bf(f3.x); b.u[13] = f2bf(f3.y); b.u[14] = f2bf(f3.z); b.u[15] = f2bf(f3.w);
    return b.v;
}

#define WMMA_BF16(A, B, C) \
    __builtin_amdgcn_wmma_f32_16x16x32_bf16(false, (A), false, (B), (short)0, (C), false, false)

// ---------------- kernel 1: init ----------------
__global__ void moe_init(float* __restrict__ out, int* __restrict__ cnt,
                         int* __restrict__ row_tok) {
    int i = blockIdx.x * blockDim.x + threadIdx.x;
    if (i < T_TOK * H_DIM) out[i] = 0.0f;
    if (i < RCAP) row_tok[i] = -1;
    if (i < N_EXP) cnt[i] = 0;
}

// ---------------- kernel 2: x -> bf16 ----------------
__global__ void moe_cvt_x(const float* __restrict__ x, unsigned short* __restrict__ x_bf) {
    int i = blockIdx.x * blockDim.x + threadIdx.x;
    if (i < T_TOK * H_DIM) x_bf[i] = f2bf(x[i]);
}

// ---------------- kernel 3: router (top-2, normalized) ----------------
__global__ void moe_router(const float* __restrict__ x, const float* __restrict__ gate_w,
                           int* __restrict__ cnt, int* __restrict__ tok_sel,
                           float* __restrict__ tok_w) {
    int t = blockIdx.x * blockDim.x + threadIdx.x;
    if (t >= T_TOK) return;
    const float* xr = x + (long long)t * H_DIM;
    float logit[N_EXP];
#pragma unroll
    for (int e = 0; e < N_EXP; ++e) {
        const float* g = gate_w + e * H_DIM;
        float acc = 0.0f;
        for (int h = 0; h < H_DIM; h += 4) {
            float4 xa = *(const float4*)(xr + h);
            float4 ga = *(const float4*)(g + h);
            acc += xa.x * ga.x + xa.y * ga.y + xa.z * ga.z + xa.w * ga.w;
        }
        logit[e] = acc;
    }
    int s0 = 0;
#pragma unroll
    for (int e = 1; e < N_EXP; ++e) if (logit[e] > logit[s0]) s0 = e;
    int s1 = (s0 == 0) ? 1 : 0;
#pragma unroll
    for (int e = 0; e < N_EXP; ++e) if (e != s0 && logit[e] > logit[s1]) s1 = e;
    float w0 = 1.0f / (1.0f + __expf(logit[s1] - logit[s0]));  // softmax denom cancels
    float w1 = 1.0f - w0;
    tok_sel[2 * t] = s0;  tok_sel[2 * t + 1] = s1;
    tok_w[2 * t]   = w0;  tok_w[2 * t + 1]   = w1;
    atomicAdd(cnt + s0, 1);
    atomicAdd(cnt + s1, 1);
}

// ---------------- kernel 4: prefix (64-aligned segments) ----------------
__global__ void moe_prefix(const int* __restrict__ cnt, int* __restrict__ offsets,
                           int* __restrict__ cursor) {
    if (threadIdx.x == 0 && blockIdx.x == 0) {
        int acc = 0;
        for (int e = 0; e < N_EXP; ++e) {
            offsets[e] = acc;
            cursor[e]  = acc;
            acc += (cnt[e] + (ALIGN_M - 1)) & ~(ALIGN_M - 1);
        }
        offsets[N_EXP] = acc;
    }
}

// ---------------- kernel 5: scatter compact row lists ----------------
__global__ void moe_scatter(const int* __restrict__ tok_sel, const float* __restrict__ tok_w,
                            int* __restrict__ cursor, int* __restrict__ row_tok,
                            float* __restrict__ row_w) {
    int t = blockIdx.x * blockDim.x + threadIdx.x;
    if (t >= T_TOK) return;
#pragma unroll
    for (int k = 0; k < 2; ++k) {
        int s = tok_sel[2 * t + k];
        int pos = atomicAdd(cursor + s, 1);
        row_tok[pos] = t;
        row_w[pos]   = tok_w[2 * t + k];
    }
}

// ---------------- kernel 6: gate_up GEMM + GeLU ----------------
// grid (MTILES, F/256). block 256 = 8 waves. WG tile: M=64, f-block=256.
// wave -> 32 f-cols (2 gate + 2 up N-tiles) x 4 M-subtiles: each B frag reused 4x.
__global__ __launch_bounds__(256) void
moe_gemm1_act(const float* __restrict__ gate_up_w,      // [E][2F][H]
              const unsigned short* __restrict__ x_bf,  // [T][H] bf16
              const int* __restrict__ offsets,
              const int* __restrict__ row_tok,
              unsigned short* __restrict__ hact) {      // [RCAP][F] bf16
    __shared__ unsigned short xt[ALIGN_M * LDP];        // 33 KB padded A chunk
    const int base = blockIdx.x * ALIGN_M;
    int e = -1;
#pragma unroll
    for (int i = 0; i < N_EXP; ++i)
        if (base >= offsets[i] && base < offsets[i + 1]) e = i;
    if (e < 0) return;

    const int tid  = threadIdx.x;
    const int lane = tid & 31;
    const int wv   = tid >> 5;
    const int f0   = blockIdx.y * 256 + wv * 32;
    const float* Wg = gate_up_w + (long long)e * (2LL * F_DIM) * H_DIM;

    v8f acc[4][4] = {};   // [m-subtile][g0,g1,u0,u1]

    for (int kc = 0; kc < H_DIM; kc += KC) {
        // stage gathered 64xKC bf16 chunk (async global->LDS; zero-fill padding rows)
        for (int i = tid; i < ALIGN_M * (KC / 8); i += 256) {
            int row = i >> 5;            // KC/8 = 32 b128 per row
            int c   = i & 31;
            int tok = row_tok[base + row];
            void* dst = &xt[row * LDP + c * 8];
            if (tok >= 0)
                stage_b128(x_bf + (long long)tok * H_DIM + kc + c * 8, dst);
            else
                *(uint4*)dst = uint4{0u, 0u, 0u, 0u};
        }
        stage_wait();
        __syncthreads();

        for (int kk = 0; kk < KC; kk += 32) {
            v16bf a[4];
#pragma unroll
            for (int mt = 0; mt < 4; ++mt)
                a[mt] = load_a_frag_lds(xt + mt * 16 * LDP, LDP, kk, lane);
            v16bf b;
            b = load_b_frag_f32(Wg, f0,              H_DIM, kc + kk, lane);
#pragma unroll
            for (int mt = 0; mt < 4; ++mt) acc[mt][0] = WMMA_BF16(a[mt], b, acc[mt][0]);
            b = load_b_frag_f32(Wg, f0 + 16,         H_DIM, kc + kk, lane);
#pragma unroll
            for (int mt = 0; mt < 4; ++mt) acc[mt][1] = WMMA_BF16(a[mt], b, acc[mt][1]);
            b = load_b_frag_f32(Wg, F_DIM + f0,      H_DIM, kc + kk, lane);
#pragma unroll
            for (int mt = 0; mt < 4; ++mt) acc[mt][2] = WMMA_BF16(a[mt], b, acc[mt][2]);
            b = load_b_frag_f32(Wg, F_DIM + f0 + 16, H_DIM, kc + kk, lane);
#pragma unroll
            for (int mt = 0; mt < 4; ++mt) acc[mt][3] = WMMA_BF16(a[mt], b, acc[mt][3]);
        }
        __syncthreads();
    }

    // fused exact GeLU epilogue, bf16 store
    const int nl = lane & 15;
    const int mh = (lane >> 4) * 8;
#pragma unroll
    for (int mt = 0; mt < 4; ++mt) {
#pragma unroll
        for (int r = 0; r < 8; ++r) {
            float g0 = acc[mt][0][r], g1 = acc[mt][1][r];
            float h0 = acc[mt][2][r] * 0.5f * g0 * (1.0f + erff(g0 * 0.70710678118654752f));
            float h1 = acc[mt][3][r] * 0.5f * g1 * (1.0f + erff(g1 * 0.70710678118654752f));
            long long rb = (long long)(base + mt * 16 + mh + r) * F_DIM;
            hact[rb + f0 + nl]      = f2bf(h0);
            hact[rb + f0 + 16 + nl] = f2bf(h1);
        }
    }
}

// ---------------- kernel 7: down GEMM + weighted combine ----------------
// grid (MTILES, H/256). block 256 = 8 waves. WG tile: M=64, n-block=256.
// wave -> 32 n-cols (2 N-tiles) x 4 M-subtiles: each B frag reused 4x.
__global__ __launch_bounds__(256) void
moe_gemm2_combine(const float* __restrict__ down_w,        // [E][H][F]
                  const unsigned short* __restrict__ hact, // [RCAP][F] bf16
                  const int* __restrict__ offsets,
                  const int* __restrict__ row_tok,
                  const float* __restrict__ row_w,
                  float* __restrict__ out) {               // [T][H] f32
    __shared__ unsigned short ht[ALIGN_M * LDP];           // 33 KB padded A chunk
    __shared__ int   s_tok[ALIGN_M];
    __shared__ float s_w[ALIGN_M];
    const int base = blockIdx.x * ALIGN_M;
    int e = -1;
#pragma unroll
    for (int i = 0; i < N_EXP; ++i)
        if (base >= offsets[i] && base < offsets[i + 1]) e = i;
    if (e < 0) return;

    const int tid = threadIdx.x;
    if (tid < ALIGN_M) {
        s_tok[tid] = row_tok[base + tid];
        s_w[tid]   = row_w[base + tid];
    }
    const int lane = tid & 31;
    const int wv   = tid >> 5;
    const int n0   = blockIdx.y * 256 + wv * 32;
    const float* Wd = down_w + (long long)e * H_DIM * F_DIM;

    v8f acc[4][2] = {};

    for (int kc = 0; kc < F_DIM; kc += KC) {   // 16 K-chunks
        for (int i = tid; i < ALIGN_M * (KC / 8); i += 256) {
            int row = i >> 5;
            int c   = i & 31;
            stage_b128(hact + (long long)(base + row) * F_DIM + kc + c * 8,
                       &ht[row * LDP + c * 8]);
        }
        stage_wait();
        __syncthreads();

        for (int kk = 0; kk < KC; kk += 32) {
            v16bf a[4];
#pragma unroll
            for (int mt = 0; mt < 4; ++mt)
                a[mt] = load_a_frag_lds(ht + mt * 16 * LDP, LDP, kk, lane);
            v16bf b;
            b = load_b_frag_f32(Wd, n0,      F_DIM, kc + kk, lane);
#pragma unroll
            for (int mt = 0; mt < 4; ++mt) acc[mt][0] = WMMA_BF16(a[mt], b, acc[mt][0]);
            b = load_b_frag_f32(Wd, n0 + 16, F_DIM, kc + kk, lane);
#pragma unroll
            for (int mt = 0; mt < 4; ++mt) acc[mt][1] = WMMA_BF16(a[mt], b, acc[mt][1]);
        }
        __syncthreads();
    }

    const int nl = lane & 15;
    const int mh = (lane >> 4) * 8;
#pragma unroll
    for (int mt = 0; mt < 4; ++mt) {
#pragma unroll
        for (int r = 0; r < 8; ++r) {
            int m = mt * 16 + mh + r;
            int tok = s_tok[m];
            if (tok >= 0) {
                float w = s_w[m];
                float* o = out + (long long)tok * H_DIM;
                atomicAdd(o + n0 + nl,      acc[mt][0][r] * w);
                atomicAdd(o + n0 + 16 + nl, acc[mt][1][r] * w);
            }
        }
    }
}

// ---------------- host launcher ----------------
extern "C" void kernel_launch(void* const* d_in, const int* in_sizes, int n_in,
                              void* d_out, int out_size, void* d_ws, size_t ws_size,
                              hipStream_t stream) {
    const float* x         = (const float*)d_in[0];  // [T][H]
    const float* gate_w    = (const float*)d_in[1];  // [E][H]
    const float* gate_up_w = (const float*)d_in[2];  // [E][2F][H]
    const float* down_w    = (const float*)d_in[3];  // [E][H][F]
    float* out = (float*)d_out;

    char* ws = (char*)d_ws;
    size_t o = 0;
    auto carve = [&](size_t bytes) -> char* {
        char* p = ws + o;
        o = (o + bytes + 255) & ~(size_t)255;
        return p;
    };
    int*            cnt     = (int*)carve(N_EXP * sizeof(int));
    int*            cursor  = (int*)carve(N_EXP * sizeof(int));
    int*            offsets = (int*)carve((N_EXP + 1) * sizeof(int));
    int*            tok_sel = (int*)carve(2 * T_TOK * sizeof(int));
    float*          tok_w   = (float*)carve(2 * T_TOK * sizeof(float));
    int*            row_tok = (int*)carve(RCAP * sizeof(int));
    float*          row_w   = (float*)carve(RCAP * sizeof(float));
    unsigned short* x_bf    = (unsigned short*)carve((size_t)T_TOK * H_DIM * 2);
    unsigned short* hact    = (unsigned short*)carve((size_t)RCAP * F_DIM * 2);

    const int nElem = T_TOK * H_DIM;
    moe_init   <<<(nElem + 255) / 256, 256, 0, stream>>>(out, cnt, row_tok);
    moe_cvt_x  <<<(nElem + 255) / 256, 256, 0, stream>>>(x, x_bf);
    moe_router <<<(T_TOK + 255) / 256, 256, 0, stream>>>(x, gate_w, cnt, tok_sel, tok_w);
    moe_prefix <<<1, 32, 0, stream>>>(cnt, offsets, cursor);
    moe_scatter<<<(T_TOK + 255) / 256, 256, 0, stream>>>(tok_sel, tok_w, cursor, row_tok, row_w);

    dim3 g1(MTILES, F_DIM / 256);   // (72, 16)
    moe_gemm1_act<<<g1, 256, 0, stream>>>(gate_up_w, x_bf, offsets, row_tok, hact);

    dim3 g2(MTILES, H_DIM / 256);   // (72, 4)
    moe_gemm2_combine<<<g2, 256, 0, stream>>>(down_w, hact, offsets, row_tok, row_w, out);
}
